// autocorrelation_block_549755814122
// MI455X (gfx1250) — matrix operand order repaired and checked by reading, not compiled, not run on hardware
//
#include <hip/hip_runtime.h>
#include <hip/hip_bf16.h>

// ---------------------------------------------------------------------------
// Autocorrelation block, MI455X (gfx1250), wave32 + WMMA bf16 + async LDS.
//
// corr[tau] = (1/1024) sum_t <q[t], k[(t-tau) mod L]>   (L=8192, d=1024)
//           = sum of wrapped diagonals of S = Q K^T  -> v_wmma_f32_16x16x32_bf16,
//             128x128 macrotiles, double-buffered LDS fed by async global->LDS.
// Then top-18 + softmax, Rp[j] = sum_k w_k * v_flat[idx_k + j], output is
// Rp tiled 1024x (the reference's flat %L gather makes it periodic).
// ---------------------------------------------------------------------------

typedef unsigned short u16;
typedef unsigned int   u32;
typedef __attribute__((ext_vector_type(16))) __bf16 v16bf;
typedef __attribute__((ext_vector_type(8)))  __bf16 bf16x8;
typedef __attribute__((ext_vector_type(8)))  float  v8f;
typedef __attribute__((ext_vector_type(4)))  int    v4i;

#define LSEQ   8192
#define DMODEL 1024
#define KTOP   18

#if defined(__gfx1250__) && __has_builtin(__builtin_amdgcn_global_load_async_to_lds_b128) && __has_builtin(__builtin_amdgcn_s_wait_asynccnt)
#define USE_ASYNC_LDS 1
#else
#define USE_ASYNC_LDS 0
#endif

#if USE_ASYNC_LDS
// Builtin signature (from clang diagnostic): param0 = v4i32 addrspace(1)*,
// param1 = LDS dest, then imm offset, imm cpol.
#define GAS(p) ((__attribute__((address_space(1))) v4i*)(p))
#define LAS(p) ((__attribute__((address_space(3))) v4i*)(p))
#endif

// ---------------- kernel 1: fp32 -> bf16 (RNE) + zero corr -----------------
__global__ __launch_bounds__(256) void cvt_kernel(const float* __restrict__ q,
                                                  const float* __restrict__ k,
                                                  u16* __restrict__ Qh,
                                                  u16* __restrict__ Kh,
                                                  float* __restrict__ corr) {
    int i = blockIdx.x * 256 + threadIdx.x;            // 8M threads
    u32 qb = __float_as_uint(q[i]);
    Qh[i] = (u16)((qb + 0x7FFFu + ((qb >> 16) & 1u)) >> 16);
    u32 kb = __float_as_uint(k[i]);
    Kh[i] = (u16)((kb + 0x7FFFu + ((kb >> 16) & 1u)) >> 16);
    if (i < LSEQ) corr[i] = 0.0f;
}

// ------------- kernel 2: WMMA GEMM + wrapped-diagonal accumulate -----------
// Block = 256 threads = 8 waves. Macrotile: 128 (t) x 128 (s).
// Wave w owns the 16-row strip [w*16, w*16+16) x all 128 cols (8 accum tiles).
__global__ __launch_bounds__(256) void corr_kernel(const u16* __restrict__ Qh,
                                                   const u16* __restrict__ Kh,
                                                   float* __restrict__ corr) {
    // +8 u16 pad per row: 80B row stride keeps 16B alignment, spreads banks.
    __shared__ __align__(16) u16 Qs[2][128][40];
    __shared__ __align__(16) u16 Ks[2][128][40];
    __shared__ float bins[256];

    const int tid  = threadIdx.x;
    const int wave = tid >> 5;
    const int lane = tid & 31;
    const int t0 = blockIdx.x * 128;
    const int s0 = blockIdx.y * 128;

    bins[tid] = 0.0f;

    v8f acc[8] = {};

    const int r = tid >> 1;        // 0..127 : row this thread stages
    const int h = tid & 1;         // 0/1    : which 16-channel half
    const u16* gq_base = Qh + (size_t)(t0 + r) * DMODEL + h * 16;
    const u16* gk_base = Kh + (size_t)(s0 + r) * DMODEL + h * 16;

    const int m    = lane & 15;
    const int off8 = (lane >> 4) * 16;
    const int boff = (lane >> 4) * 16;

#if USE_ASYNC_LDS
    // ---- double-buffered pipeline: async global->LDS (ASYNCcnt) ----
    {
        u16* lq = &Qs[0][r][h * 16];
        u16* lk = &Ks[0][r][h * 16];
        __builtin_amdgcn_global_load_async_to_lds_b128(GAS(gq_base),     LAS(lq),     0, 0);
        __builtin_amdgcn_global_load_async_to_lds_b128(GAS(gq_base + 8), LAS(lq + 8), 0, 0);
        __builtin_amdgcn_global_load_async_to_lds_b128(GAS(gk_base),     LAS(lk),     0, 0);
        __builtin_amdgcn_global_load_async_to_lds_b128(GAS(gk_base + 8), LAS(lk + 8), 0, 0);
    }
    for (int step = 0; step < DMODEL / 32; ++step) {
        const int cb = step & 1;
        if (step + 1 < DMODEL / 32) {
            const int c1 = (step + 1) * 32;
            u16* lq = &Qs[cb ^ 1][r][h * 16];
            u16* lk = &Ks[cb ^ 1][r][h * 16];
            __builtin_amdgcn_global_load_async_to_lds_b128(GAS(gq_base + c1),     LAS(lq),     0, 0);
            __builtin_amdgcn_global_load_async_to_lds_b128(GAS(gq_base + c1 + 8), LAS(lq + 8), 0, 0);
            __builtin_amdgcn_global_load_async_to_lds_b128(GAS(gk_base + c1),     LAS(lk),     0, 0);
            __builtin_amdgcn_global_load_async_to_lds_b128(GAS(gk_base + c1 + 8), LAS(lk + 8), 0, 0);
            __builtin_amdgcn_s_wait_asynccnt((unsigned short)4);   // current step's 4 done (in-order)
        } else {
            __builtin_amdgcn_s_wait_asynccnt((unsigned short)0);
        }
        __syncthreads();                 // everyone's current-buffer writes visible

        // A fragment (16x32): two contiguous 8-elem chunks (contiguous k = 16*half + s, same order as B).
        bf16x8 a0 = *(const bf16x8*)&Qs[cb][wave * 16 + m][off8];
        bf16x8 a1 = *(const bf16x8*)&Qs[cb][wave * 16 + m][off8 + 8];
        v16bf a;
        #pragma unroll
        for (int j = 0; j < 8; ++j) { a[j] = a0[j]; a[8 + j] = a1[j]; }

        // Batch all 8 B fragments first so dscnt waits overlap the WMMA chain.
        v16bf bf[8];
        #pragma unroll
        for (int n = 0; n < 8; ++n) {
            bf16x8 b0 = *(const bf16x8*)&Ks[cb][n * 16 + m][boff + 0];
            bf16x8 b1 = *(const bf16x8*)&Ks[cb][n * 16 + m][boff + 8];
            #pragma unroll
            for (int j = 0; j < 8; ++j) { bf[n][j] = b0[j]; bf[n][8 + j] = b1[j]; }
        }
        #pragma unroll
        for (int n = 0; n < 8; ++n)
            acc[n] = __builtin_amdgcn_wmma_f32_16x16x32_bf16(
                false, a, false, bf[n], (short)0, acc[n], false, false);

        __syncthreads();                 // buffer cb free for step+2's prefetch
    }
#else
    // ---- fallback: synchronous staging, single buffer ----
    for (int c0 = 0; c0 < DMODEL; c0 += 32) {
        __syncthreads();
        const uint4* qsrc = (const uint4*)(gq_base + c0);
        const uint4* ksrc = (const uint4*)(gk_base + c0);
        uint4 q0 = qsrc[0], q1 = qsrc[1];
        uint4 k0 = ksrc[0], k1 = ksrc[1];
        *(uint4*)&Qs[0][r][h * 16 + 0] = q0;
        *(uint4*)&Qs[0][r][h * 16 + 8] = q1;
        *(uint4*)&Ks[0][r][h * 16 + 0] = k0;
        *(uint4*)&Ks[0][r][h * 16 + 8] = k1;
        __syncthreads();

        bf16x8 a0 = *(const bf16x8*)&Qs[0][wave * 16 + m][off8];
        bf16x8 a1 = *(const bf16x8*)&Qs[0][wave * 16 + m][off8 + 8];
        v16bf a;
        #pragma unroll
        for (int j = 0; j < 8; ++j) { a[j] = a0[j]; a[8 + j] = a1[j]; }

        v16bf bf[8];
        #pragma unroll
        for (int n = 0; n < 8; ++n) {
            bf16x8 b0 = *(const bf16x8*)&Ks[0][n * 16 + m][boff + 0];
            bf16x8 b1 = *(const bf16x8*)&Ks[0][n * 16 + m][boff + 8];
            #pragma unroll
            for (int j = 0; j < 8; ++j) { bf[n][j] = b0[j]; bf[n][8 + j] = b1[j]; }
        }
        #pragma unroll
        for (int n = 0; n < 8; ++n)
            acc[n] = __builtin_amdgcn_wmma_f32_16x16x32_bf16(
                false, a, false, bf[n], (short)0, acc[n], false, false);
    }
    __syncthreads();
#endif

    // D layout: VGPR rr -> M = rr + (lane>=16 ? 8 : 0), N = lane%16.
    const int hi = (lane >> 4) * 8;
    #pragma unroll
    for (int n = 0; n < 8; ++n) {
        const int ds = n * 16 + m;                 // local s
        #pragma unroll
        for (int rr = 0; rr < 8; ++rr) {
            const int dt  = wave * 16 + rr + hi;   // local t
            const int bin = dt - ds + 127;         // 0..254
            __hip_atomic_fetch_add(&bins[bin], acc[n][rr],
                                   __ATOMIC_RELAXED, __HIP_MEMORY_SCOPE_WORKGROUP);
        }
    }
    __syncthreads();

    if (tid < 255) {
        int tau = (t0 - s0 + (tid - 127)) & (LSEQ - 1);   // mod 8192 (2's comp ok)
        __hip_atomic_fetch_add(&corr[tau], bins[tid],
                               __ATOMIC_RELAXED, __HIP_MEMORY_SCOPE_AGENT);
    }
}

// ---------------- kernel 3: top-18 + softmax (single block) ----------------
__global__ __launch_bounds__(256) void topk_kernel(const float* __restrict__ corr,
                                                   float* __restrict__ wout,
                                                   int* __restrict__ iout) {
    __shared__ float sc[LSEQ];
    __shared__ float rv[256];
    __shared__ int   ri[256];
    __shared__ float vals[KTOP];
    __shared__ int   idxs[KTOP];
    const int tid = threadIdx.x;
    for (int i = tid; i < LSEQ; i += 256) sc[i] = corr[i] * (1.0f / 1024.0f);
    __syncthreads();
    for (int k = 0; k < KTOP; ++k) {
        float bv = -__builtin_inff();
        int   bi = 0;
        for (int i = tid; i < LSEQ; i += 256)
            if (sc[i] > bv) { bv = sc[i]; bi = i; }
        rv[tid] = bv; ri[tid] = bi;
        __syncthreads();
        for (int off = 128; off > 0; off >>= 1) {
            if (tid < off && rv[tid + off] > rv[tid]) {
                rv[tid] = rv[tid + off]; ri[tid] = ri[tid + off];
            }
            __syncthreads();
        }
        if (tid == 0) {
            vals[k] = rv[0]; idxs[k] = ri[0];
            sc[ri[0]] = -__builtin_inff();
        }
        __syncthreads();
    }
    if (tid == 0) {
        float mx = vals[0];               // iterative argmax -> vals[0] is max
        float e[KTOP], s = 0.0f;
        for (int k = 0; k < KTOP; ++k) { e[k] = __expf(vals[k] - mx); s += e[k]; }
        float inv = 1.0f / s;
        for (int k = 0; k < KTOP; ++k) { wout[k] = e[k] * inv; iout[k] = idxs[k]; }
    }
}

// ---------------- kernel 4: Rp[j] = sum_k w_k * v_flat[idx_k + j] ----------
__global__ __launch_bounds__(256) void rp_kernel(const float* __restrict__ v,
                                                 const float* __restrict__ w,
                                                 const int* __restrict__ idx,
                                                 float* __restrict__ Rp) {
    __shared__ float wl[KTOP];
    __shared__ int   il[KTOP];
    if (threadIdx.x < KTOP) { wl[threadIdx.x] = w[threadIdx.x]; il[threadIdx.x] = idx[threadIdx.x]; }
    __syncthreads();
    int j = blockIdx.x * 256 + threadIdx.x;            // 0..8191
    float r = 0.0f;
    #pragma unroll
    for (int k = 0; k < KTOP; ++k) r += wl[k] * v[il[k] + j];   // idx+j <= 16382 < L*d
    Rp[j] = r;
}

// ---------------- kernel 5: broadcast Rp tiled 1024x (32MB write) ----------
__global__ __launch_bounds__(256) void bcast_kernel(const float4* __restrict__ Rp4,
                                                    float4* __restrict__ out4) {
    int i = blockIdx.x * 256 + threadIdx.x;            // 2M float4s
    out4[i] = Rp4[i & 2047];
}

// ---------------------------------------------------------------------------
extern "C" void kernel_launch(void* const* d_in, const int* in_sizes, int n_in,
                              void* d_out, int out_size, void* d_ws, size_t ws_size,
                              hipStream_t stream) {
    const float* q = (const float*)d_in[0];
    const float* k = (const float*)d_in[1];
    const float* v = (const float*)d_in[2];
    float* out = (float*)d_out;

    // workspace layout (needs ~33.6 MB)
    const size_t QK_BYTES = (size_t)LSEQ * DMODEL * sizeof(u16);   // 16 MB each
    char* ws = (char*)d_ws;
    u16*   Qh   = (u16*)ws;
    u16*   Kh   = (u16*)(ws + QK_BYTES);
    float* corr = (float*)(ws + 2 * QK_BYTES);
    float* Rp   = (float*)(ws + 2 * QK_BYTES + 32768);
    float* wts  = (float*)(ws + 2 * QK_BYTES + 65536);
    int*   idxs = (int*)  (ws + 2 * QK_BYTES + 65536 + 128);

    cvt_kernel  <<<32768, 256, 0, stream>>>(q, k, Qh, Kh, corr);
    corr_kernel <<<dim3(64, 64), 256, 0, stream>>>(Qh, Kh, corr);
    topk_kernel <<<1, 256, 0, stream>>>(corr, wts, idxs);
    rp_kernel   <<<32, 256, 0, stream>>>(v, wts, idxs, Rp);
    bcast_kernel<<<8192, 256, 0, stream>>>((const float4*)Rp, (float4*)out);
}
